// ANEDecoderLayer_9191230013799
// MI455X (gfx1250) — matrix operand order repaired and checked by reading, not compiled, not run on hardware
//
#include <hip/hip_runtime.h>
#include <cstdint>
#include <cstddef>

// ---------------------------------------------------------------------------
// Problem constants (from the reference): B=2, C=2048, S=1024, H=16, HD=128,
// FF=8192.  All GEMM dims divide the 128x256x32 tiling exactly.
// ---------------------------------------------------------------------------
#define B_   2
#define C_   2048
#define S_   1024
#define H_   16
#define HD_  128
#define FF_  8192
#define HHD_ (H_ * HD_)          // 2048
#define EPS_   1e-5f
#define SCALE_ 0.08838834764831845f   // 1/sqrt(128)

typedef __attribute__((ext_vector_type(16))) __bf16 v16bf;
typedef __attribute__((ext_vector_type(8)))  float  v8f;

// fp32 -> bf16 via hardware fptrunc (round-to-nearest-even); lets the backend
// emit v_cvt ops instead of a 4-op integer rounding sequence.
__device__ __forceinline__ unsigned short f2bf(float f) {
  union { __bf16 h; unsigned short u; } cv;
  cv.h = (__bf16)f;
  return cv.u;
}

union FragU { v16bf v; uint4 q[2]; };

// A-matrix 16x32 bf16 fragment (ISA 7.12.2): lane L holds row (L&15);
// lane-half hi=L>>4 selects K pairs: VGPR0-3 = K hi*8..hi*8+7, VGPR4-7 = K 16+hi*8..
__device__ __forceinline__ v16bf load_fragA(const unsigned short* rowp, int hi) {
  FragU f;
  f.q[0] = *(const uint4*)(rowp + hi * 8);
  f.q[1] = *(const uint4*)(rowp + 16 + hi * 8);
  return f.v;
}
// B-matrix 32x16 bf16 fragment: lane L holds col (L&15); same K order as A:
// VGPR0-3 = K hi*8..hi*8+7, VGPR4-7 = K 16+hi*8..
__device__ __forceinline__ v16bf load_fragB(const unsigned short* rowp, int hi) {
  FragU f;
  f.q[0] = *(const uint4*)(rowp + hi * 8);
  f.q[1] = *(const uint4*)(rowp + 16 + hi * 8);
  return f.v;
}

// ---------------------------------------------------------------------------
// NT WMMA GEMM: Out[z][m][n] = sum_k A[z][m][k] * B[z][n][k]
//   A: fp32 (converted to bf16 while staging) when A_F32, else bf16.
//   B: always bf16, K-contiguous rows.
//   OUTM: 0 = fp32 out, 1 = bf16 out, 2 = fp32 out + residual add.
// Block: 256 threads (8 wave32 waves), tile 128(M) x 256(N), K-slab 32.
// Each wave: 64x64 tile = 4x4 v_wmma_f32_16x16x32_bf16 accumulators
// (16 WMMA per slab vs 16 ds_load_b128 -> 1:1 mix, ~128 acc VGPRs).
// ---------------------------------------------------------------------------
template<bool A_F32, int OUTM>
__global__ __launch_bounds__(256) void gemm_nt(
    const void* __restrict__ Av, const unsigned short* __restrict__ Bm,
    void* __restrict__ Out, const float* __restrict__ Res,
    int M, int N, int K,
    long aBatch, long bBatch, long cBatch, long rBatch)
{
  (void)M;
  __shared__ unsigned short sA[128][40];   // 40 = 32 + 8 pad (80B rows, 16B aligned)
  __shared__ unsigned short sB[256][40];

  const int tid = threadIdx.x;
  const int z   = blockIdx.z;
  const int m0  = blockIdx.x * 128;
  const int n0  = blockIdx.y * 256;

  const float*          Af = (const float*)Av + z * aBatch;
  const unsigned short* Ab = (const unsigned short*)Av + z * aBatch;
  const unsigned short* Bp = Bm + z * bBatch;

  const int lane  = tid & 31;
  const int wid   = tid >> 5;
  const int wm    = (wid & 1) * 64;       // wave row offset in tile
  const int wn    = (wid >> 1) * 64;      // wave col offset in tile
  const int row16 = lane & 15;
  const int hi    = (lane >> 4) & 1;

  const v8f vzero = {0.f, 0.f, 0.f, 0.f, 0.f, 0.f, 0.f, 0.f};
  v8f acc[4][4];
#pragma unroll
  for (int i = 0; i < 4; ++i)
#pragma unroll
    for (int j = 0; j < 4; ++j) acc[i][j] = vzero;

  const int sr = tid >> 3;          // staging row 0..31
  const int c4 = (tid & 7) << 2;    // staging K offset 0..28

  for (int k0 = 0; k0 < K; k0 += 32) {
#pragma unroll
    for (int j = 0; j < 8; ++j) {
      const int rr = sr + j * 32;
      if (j < 4) {                       // A tile: 128 rows
        const long ga = (long)(m0 + rr) * K + k0 + c4;
        if (A_F32) {
          const float4 w4 = *(const float4*)(Af + ga);
          sA[rr][c4 + 0] = f2bf(w4.x);
          sA[rr][c4 + 1] = f2bf(w4.y);
          sA[rr][c4 + 2] = f2bf(w4.z);
          sA[rr][c4 + 3] = f2bf(w4.w);
        } else {
          *(uint2*)&sA[rr][c4] = *(const uint2*)(Ab + ga);
        }
      }
      const long gb = (long)(n0 + rr) * K + k0 + c4;   // B tile: 256 rows
      *(uint2*)&sB[rr][c4] = *(const uint2*)(Bp + gb);
    }
    if (k0 + 32 < K) {  // uniform branch; CDNA5 global_prefetch_b8 of next K slab
      __builtin_prefetch((const char*)(Bp + ((long)(n0 + sr) * K + k0 + 32 + c4)), 0, 1);
      if (A_F32)
        __builtin_prefetch((const char*)(Af + ((long)(m0 + sr) * K + k0 + 32 + c4)), 0, 1);
    }
    __syncthreads();

    v16bf afr[4];
#pragma unroll
    for (int tm = 0; tm < 4; ++tm)
      afr[tm] = load_fragA(&sA[wm + tm * 16 + row16][0], hi);
#pragma unroll
    for (int tn = 0; tn < 4; ++tn) {
      const v16bf bfr = load_fragB(&sB[wn + tn * 16 + row16][0], hi);
#pragma unroll
      for (int tm = 0; tm < 4; ++tm)
        acc[tm][tn] = __builtin_amdgcn_wmma_f32_16x16x32_bf16(
            false, afr[tm], false, bfr, (short)0, acc[tm][tn], false, false);
    }
    __syncthreads();
  }

  // Epilogue: C/D layout (ISA 7.12.2): VGPR r, lanes 0-15 -> (M=r, N=lane),
  // lanes 16-31 -> (M=8+r, N=lane-16).
#pragma unroll
  for (int tm = 0; tm < 4; ++tm) {
#pragma unroll
    for (int tn = 0; tn < 4; ++tn) {
      const v8f a  = acc[tm][tn];
      const int mb = m0 + wm + tm * 16 + hi * 8;
      const int nn = n0 + wn + tn * 16 + row16;
#pragma unroll
      for (int r = 0; r < 8; ++r) {
        const long ci = cBatch * z + (long)(mb + r) * N + nn;
        const float vv = a[r];
        if (OUTM == 0) {
          ((float*)Out)[ci] = vv;
        } else if (OUTM == 1) {
          ((unsigned short*)Out)[ci] = f2bf(vv);
        } else {
          const long ri = rBatch * z + (long)(mb + r) * N + nn;
          ((float*)Out)[ci] = vv + Res[ri];
        }
      }
    }
  }
}

// ---------------------------------------------------------------------------
// RMSNorm over channel dim of x (B,C,S); emit bf16 in (B,S,C) K-major layout.
// grid (S/32, B), block (32,8).
// ---------------------------------------------------------------------------
__global__ void rmsnorm_kernel(const float* __restrict__ x, const float* __restrict__ w,
                               unsigned short* __restrict__ out) {
  const int tx = threadIdx.x, ty = threadIdx.y;
  const int s = blockIdx.x * 32 + tx;
  const int b = blockIdx.y;
  float sum = 0.f;
  for (int c = ty; c < C_; c += 8) {
    const float v = x[((long)b * C_ + c) * S_ + s];
    sum += v * v;
  }
  __shared__ float red[8][32];
  red[ty][tx] = sum;
  __syncthreads();
  if (ty == 0) {
    float t = 0.f;
#pragma unroll
    for (int j = 0; j < 8; ++j) t += red[j][tx];
    red[0][tx] = rsqrtf(t / (float)C_ + EPS_);
  }
  __syncthreads();
  const float rinv = red[0][tx];
  for (int c = ty; c < C_; c += 8) {
    const float v = x[((long)b * C_ + c) * S_ + s] * rinv * w[c];
    out[((long)b * S_ + s) * C_ + c] = f2bf(v);
  }
}

// ---------------------------------------------------------------------------
// RoPE (rotate-half over head dim) + transpose (B,O,S) f32 -> (B,H,S,HD) bf16.
// One thread per element; d is fastest so writes are fully coalesced.
// ---------------------------------------------------------------------------
__global__ void rope_kernel(const float* __restrict__ p, const float* __restrict__ sn,
                            const float* __restrict__ cs, unsigned short* __restrict__ out) {
  const long idx = (long)blockIdx.x * 256 + threadIdx.x;   // (hb, s, d)
  const int d  = (int)(idx & (HD_ - 1));
  const int s  = (int)((idx >> 7) & (S_ - 1));
  const long hb = idx >> 17;                                // b*H + h
  const float xv = p[(hb * HD_ + d) * (long)S_ + s];
  const int d2 = (d < HD_ / 2) ? d + HD_ / 2 : d - HD_ / 2;
  float rv = p[(hb * HD_ + d2) * (long)S_ + s];
  if (d < HD_ / 2) rv = -rv;
  out[idx] = f2bf(xv * cs[d * S_ + s] + rv * sn[d * S_ + s]);
}

// ---------------------------------------------------------------------------
// Softmax over k of scores_t (B,H,Sq,Sk) with scale + additive mask(k,q);
// emits bf16 probs_t.  Block per (z, q), 256 threads, 4 k each.
// ---------------------------------------------------------------------------
__global__ __launch_bounds__(256) void softmax_kernel(const float* __restrict__ sc,
                                                      const float* __restrict__ mask,
                                                      unsigned short* __restrict__ pr) {
  const int q = blockIdx.x;
  const long z = blockIdx.y;
  const int tid = threadIdx.x;
  const long base = (z * S_ + q) * (long)S_;
  float v[4];
  float mx = -3.0e38f;
#pragma unroll
  for (int i = 0; i < 4; ++i) {
    const int k = tid + i * 256;
    v[i] = sc[base + k] * SCALE_ + mask[(long)k * S_ + q];
    mx = fmaxf(mx, v[i]);
  }
  __shared__ float red[256];
  red[tid] = mx;
  __syncthreads();
  for (int o = 128; o > 0; o >>= 1) {
    if (tid < o) red[tid] = fmaxf(red[tid], red[tid + o]);
    __syncthreads();
  }
  mx = red[0];
  __syncthreads();
  float sme = 0.f;
#pragma unroll
  for (int i = 0; i < 4; ++i) { v[i] = __expf(v[i] - mx); sme += v[i]; }
  red[tid] = sme;
  __syncthreads();
  for (int o = 128; o > 0; o >>= 1) {
    if (tid < o) red[tid] += red[tid + o];
    __syncthreads();
  }
  const float inv = 1.f / red[0];
#pragma unroll
  for (int i = 0; i < 4; ++i) pr[base + tid + i * 256] = f2bf(v[i] * inv);
}

// ---------------------------------------------------------------------------
// LDS-tiled transpose + f32->bf16: in (z,R,Cc) f32 -> out (z,Cc,R) bf16.
// grid (Cc/32, R/32, z), block (32,8).
// ---------------------------------------------------------------------------
__global__ void transpose_cvt_kernel(const float* __restrict__ in, unsigned short* __restrict__ out,
                                     int R, int Cc, long inB, long outB) {
  __shared__ float t[32][33];
  const int tx = threadIdx.x, ty = threadIdx.y;
  const int c0 = blockIdx.x * 32, r0 = blockIdx.y * 32;
  const long ib = inB * blockIdx.z, ob = outB * blockIdx.z;
#pragma unroll
  for (int j = 0; j < 4; ++j)
    t[ty + j * 8][tx] = in[ib + (long)(r0 + ty + j * 8) * Cc + c0 + tx];
  __syncthreads();
#pragma unroll
  for (int j = 0; j < 4; ++j)
    out[ob + (long)(c0 + ty + j * 8) * R + r0 + tx] = f2bf(t[tx][ty + j * 8]);
}

// ---------------------------------------------------------------------------
// silu(gate)*up fused with transpose + bf16: (z,R=FF,Cc=S) -> (z,S,FF) bf16.
// ---------------------------------------------------------------------------
__global__ void silu_mul_t_kernel(const float* __restrict__ g, const float* __restrict__ u,
                                  unsigned short* __restrict__ out, int R, int Cc,
                                  long inB, long outB) {
  __shared__ float t[32][33];
  const int tx = threadIdx.x, ty = threadIdx.y;
  const int c0 = blockIdx.x * 32, r0 = blockIdx.y * 32;
  const long ib = inB * blockIdx.z, ob = outB * blockIdx.z;
#pragma unroll
  for (int j = 0; j < 4; ++j) {
    const long gi = ib + (long)(r0 + ty + j * 8) * Cc + c0 + tx;
    const float gv = g[gi];
    const float uv = u[gi];
    t[ty + j * 8][tx] = gv * (1.f / (1.f + __expf(-gv))) * uv;
  }
  __syncthreads();
#pragma unroll
  for (int j = 0; j < 4; ++j)
    out[ob + (long)(c0 + ty + j * 8) * R + r0 + tx] = f2bf(t[tx][ty + j * 8]);
}

// ---------------------------------------------------------------------------
// Workspace layout (bytes).  Peak ~320 MB; gate/up reuse the scores region,
// the SiLU activation reuses the probs region.
// ---------------------------------------------------------------------------
static constexpr size_t OFF_XN  = 0;                                        // (B,S,C) bf16
static constexpr size_t OFF_QP  = OFF_XN  + (size_t)B_ * S_ * C_ * 2;       // (B,O,S) f32
static constexpr size_t OFF_KP  = OFF_QP  + (size_t)B_ * HHD_ * S_ * 4;
static constexpr size_t OFF_VBF = OFF_KP  + (size_t)B_ * HHD_ * S_ * 4;     // (B,O,S) bf16
static constexpr size_t OFF_QR  = OFF_VBF + (size_t)B_ * HHD_ * S_ * 2;     // (B,H,S,HD) bf16
static constexpr size_t OFF_KR  = OFF_QR  + (size_t)B_ * H_ * S_ * HD_ * 2;
static constexpr size_t OFF_SC  = OFF_KR  + (size_t)B_ * H_ * S_ * HD_ * 2; // (B,H,Sq,Sk) f32
static constexpr size_t OFF_PR  = OFF_SC  + (size_t)B_ * H_ * S_ * S_ * 4;  // bf16
static constexpr size_t OFF_AT  = OFF_PR  + (size_t)B_ * H_ * S_ * S_ * 2;  // (B,O,S) f32
static constexpr size_t OFF_ATT = OFF_AT  + (size_t)B_ * HHD_ * S_ * 4;     // (B,S,O) bf16
static constexpr size_t OFF_X1  = OFF_ATT + (size_t)B_ * S_ * HHD_ * 2;     // (B,C,S) f32
static constexpr size_t OFF_X2  = OFF_X1  + (size_t)B_ * C_ * S_ * 4;
static constexpr size_t OFF_GATE = OFF_SC;                                  // reuse scores
static constexpr size_t OFF_UP   = OFF_SC + (size_t)B_ * FF_ * S_ * 4;      // fits in scores
static constexpr size_t OFF_ACT  = OFF_PR;                                  // reuse probs

// ---------------------------------------------------------------------------
// One attention block (self- and cross-attention share this exactly; the KV
// cache can be ignored because kv_write_idx = arange(S) fully overwrites it).
// ---------------------------------------------------------------------------
static void attention_block(const float* x_in, const float* w_rms,
                            const float* wq, const float* wk, const float* wv, const float* wo,
                            const float* sin_q, const float* cos_q,
                            const float* sin_k, const float* cos_k,
                            const float* mask, float* x_out, char* ws, hipStream_t stream) {
  unsigned short* xn    = (unsigned short*)(ws + OFF_XN);
  float*          qp    = (float*)(ws + OFF_QP);
  float*          kp    = (float*)(ws + OFF_KP);
  unsigned short* vbf   = (unsigned short*)(ws + OFF_VBF);
  unsigned short* qr    = (unsigned short*)(ws + OFF_QR);
  unsigned short* kr    = (unsigned short*)(ws + OFF_KR);
  float*          sc    = (float*)(ws + OFF_SC);
  unsigned short* pr    = (unsigned short*)(ws + OFF_PR);
  float*          attn  = (float*)(ws + OFF_AT);
  unsigned short* attnt = (unsigned short*)(ws + OFF_ATT);

  rmsnorm_kernel<<<dim3(S_ / 32, B_), dim3(32, 8), 0, stream>>>(x_in, w_rms, xn);

  const dim3 gproj(HHD_ / 128, S_ / 256, B_);
  gemm_nt<true, 0><<<gproj, 256, 0, stream>>>(wq, xn, qp, nullptr, HHD_, S_, C_,
                                              0, (long)S_ * C_, (long)HHD_ * S_, 0);
  gemm_nt<true, 0><<<gproj, 256, 0, stream>>>(wk, xn, kp, nullptr, HHD_, S_, C_,
                                              0, (long)S_ * C_, (long)HHD_ * S_, 0);
  gemm_nt<true, 1><<<gproj, 256, 0, stream>>>(wv, xn, vbf, nullptr, HHD_, S_, C_,
                                              0, (long)S_ * C_, (long)HHD_ * S_, 0);

  const int nrope = (B_ * H_ * S_ * HD_) / 256;
  rope_kernel<<<nrope, 256, 0, stream>>>(qp, sin_q, cos_q, qr);
  rope_kernel<<<nrope, 256, 0, stream>>>(kp, sin_k, cos_k, kr);

  // scores_t[z][q][k] = q . k  (NT over HD)
  gemm_nt<false, 0><<<dim3(S_ / 128, S_ / 256, B_ * H_), 256, 0, stream>>>(
      qr, kr, sc, nullptr, S_, S_, HD_,
      (long)S_ * HD_, (long)S_ * HD_, (long)S_ * S_, 0);

  softmax_kernel<<<dim3(S_, B_ * H_), 256, 0, stream>>>(sc, mask, pr);

  // attn[z][d][q] = v . probs_t  (NT over Sk)
  gemm_nt<false, 0><<<dim3(HD_ / 128, S_ / 256, B_ * H_), 256, 0, stream>>>(
      vbf, pr, attn, nullptr, HD_, S_, S_,
      (long)HD_ * S_, (long)S_ * S_, (long)HD_ * S_, 0);

  transpose_cvt_kernel<<<dim3(S_ / 32, HHD_ / 32, B_), dim3(32, 8), 0, stream>>>(
      attn, attnt, HHD_, S_, (long)HHD_ * S_, (long)S_ * HHD_);

  // x_out = x_in + Wo @ attn   (residual folded into epilogue)
  gemm_nt<true, 2><<<dim3(C_ / 128, S_ / 256, B_), 256, 0, stream>>>(
      wo, attnt, x_out, x_in, C_, S_, HHD_,
      0, (long)S_ * HHD_, (long)C_ * S_, (long)C_ * S_);
}

// ---------------------------------------------------------------------------
extern "C" void kernel_launch(void* const* d_in, const int* in_sizes, int n_in,
                              void* d_out, int out_size, void* d_ws, size_t ws_size,
                              hipStream_t stream) {
  (void)in_sizes; (void)n_in; (void)out_size; (void)ws_size;
  const float* x      = (const float*)d_in[0];
  const float* sin_q  = (const float*)d_in[1];
  const float* cos_q  = (const float*)d_in[2];
  const float* sin_k  = (const float*)d_in[3];
  const float* cos_k  = (const float*)d_in[4];
  // d_in[5] kv_write_idx, d_in[6] kv_layer_idx, d_in[9]/[10] caches: unused
  // (arange write index fully overwrites the cache, so key==k, val==v).
  const float* mself  = (const float*)d_in[7];
  const float* mcross = (const float*)d_in[8];
  const float* w_sa   = (const float*)d_in[11];
  const float* wq_sa  = (const float*)d_in[12];
  const float* wk_sa  = (const float*)d_in[13];
  const float* wv_sa  = (const float*)d_in[14];
  const float* wo_sa  = (const float*)d_in[15];
  const float* w_ca   = (const float*)d_in[16];
  const float* wq_ca  = (const float*)d_in[17];
  const float* wk_ca  = (const float*)d_in[18];
  const float* wv_ca  = (const float*)d_in[19];
  const float* wo_ca  = (const float*)d_in[20];
  const float* w_mlp  = (const float*)d_in[21];
  const float* w_gate = (const float*)d_in[22];
  const float* w_up   = (const float*)d_in[23];
  const float* w_down = (const float*)d_in[24];

  char* ws = (char*)d_ws;
  float* x1 = (float*)(ws + OFF_X1);
  float* x2 = (float*)(ws + OFF_X2);

  // Self-attention block: x1 = x + Attn(RMS(x))
  attention_block(x, w_sa, wq_sa, wk_sa, wv_sa, wo_sa,
                  sin_q, cos_q, sin_k, cos_k, mself, x1, ws, stream);

  // Cross-attention block (Xq == Xkv == x1): x2 = x1 + Attn(RMS(x1))
  attention_block(x1, w_ca, wq_ca, wk_ca, wv_ca, wo_ca,
                  sin_q, cos_q, sin_k, cos_k, mcross, x2, ws, stream);

  // Gated-SiLU MLP: out = x2 + Wdown @ (silu(Wg @ h) * Wu @ h)
  unsigned short* xn  = (unsigned short*)(ws + OFF_XN);
  float*          gate = (float*)(ws + OFF_GATE);
  float*          up   = (float*)(ws + OFF_UP);
  unsigned short* act  = (unsigned short*)(ws + OFF_ACT);

  rmsnorm_kernel<<<dim3(S_ / 32, B_), dim3(32, 8), 0, stream>>>(x2, w_mlp, xn);
  gemm_nt<true, 0><<<dim3(FF_ / 128, S_ / 256, B_), 256, 0, stream>>>(
      w_gate, xn, gate, nullptr, FF_, S_, C_, 0, (long)S_ * C_, (long)FF_ * S_, 0);
  gemm_nt<true, 0><<<dim3(FF_ / 128, S_ / 256, B_), 256, 0, stream>>>(
      w_up, xn, up, nullptr, FF_, S_, C_, 0, (long)S_ * C_, (long)FF_ * S_, 0);
  silu_mul_t_kernel<<<dim3(S_ / 32, FF_ / 32, B_), dim3(32, 8), 0, stream>>>(
      gate, up, act, FF_, S_, (long)FF_ * S_, (long)S_ * FF_);
  gemm_nt<true, 2><<<dim3(C_ / 128, S_ / 256, B_), 256, 0, stream>>>(
      w_down, act, (float*)d_out, x2, C_, S_, FF_,
      0, (long)S_ * FF_, (long)C_ * S_, (long)C_ * S_);
}
